// MAFGNN_13288628814204
// MI455X (gfx1250) — compile-verified, matrix operand order checked by reading, not run on hardware
//
#include <hip/hip_runtime.h>
#include <hip/hip_bf16.h>

typedef __attribute__((ext_vector_type(16))) __bf16 v16bf;
typedef __attribute__((ext_vector_type(8)))  float  v8f;
typedef unsigned short u16t;
typedef unsigned int   u32t;

__device__ __forceinline__ u16t f2bf(float f) {
  union { float f; u32t u; } v; v.f = f;
  u32t r = v.u + 0x7FFFu + ((v.u >> 16) & 1u);   // round-to-nearest-even
  return (u16t)(r >> 16);
}
__device__ __forceinline__ float bf2f(u16t h) {
  union { u32t u; float f; } v; v.u = ((u32t)h) << 16;
  return v.f;
}

union FragBF {
  v16bf bv;
  uint4 q[2];
  u16t  u[16];
};

// ---------------------------------------------------------------------------
// Generic bf16 WMMA GEMM:  C[z] = alpha * (A @ B[z]) + ebeta*E[z] (- I if addIdent)
// A: M x K (row major, ld=K, shared across z). B: K x NC (ld=NC, + z*bStride).
// C: M x NC bf16 (+ z*cStride). Workgroup tile 128x128, 8 waves, wave tile 64x32.
// ---------------------------------------------------------------------------
__global__ __launch_bounds__(256) void gemm_bf16_wmma(
    const u16t* __restrict__ A, const u16t* __restrict__ Bsrc,
    u16t* __restrict__ C, int K, int NC,
    long long bStride, long long cStride,
    float alpha, const u16t* __restrict__ E, float ebeta, int addIdent)
{
  __shared__ u16t sA[128][40];   // [m][k]   (pad 32->40 keeps 16B alignment)
  __shared__ u16t sB[128][40];   // [n][k]   (B tile stored transposed)

  const int tid  = threadIdx.x;
  const int lane = tid & 31;
  const int wv   = tid >> 5;
  const int wm   = wv & 1;        // 2 waves along M (64 rows each)
  const int wn   = wv >> 1;       // 4 waves along N (32 cols each)
  const int mb   = blockIdx.y * 128;
  const int nb   = blockIdx.x * 128;
  const u16t* Bp = Bsrc + (long long)blockIdx.z * bStride;
  u16t*       Cp = C    + (long long)blockIdx.z * cStride;

  const int half = lane >> 4;
  const int l16  = lane & 15;
  const int koff = half * 8;      // ISA bf16 A/B fragment K-run offsets

  v8f acc[4][2] = {};

  const int ar = tid >> 1;          // A-tile row   0..127
  const int ah = (tid & 1) * 16;    // A-tile k sub-offset 0/16
  const int bk = (tid & 15) * 2;    // B-tile k pair 0..30
  const int bn = (tid >> 4) * 8;    // B-tile n group 0..120

  const int ksteps = K >> 5;
  for (int kk = 0; kk < ksteps; ++kk) {
    const int k0 = kk << 5;
    // stage A tile (rows contiguous in K)
    {
      const u16t* gp = A + (long long)(mb + ar) * K + (k0 + ah);
      *(uint4*)&sA[ar][ah]     = *(const uint4*)(gp);
      *(uint4*)&sA[ar][ah + 8] = *(const uint4*)(gp + 8);
    }
    // stage B tile transposed: sB[n][k] = B[k0+k][nb+n]
    {
      const u16t* g0 = Bp + (long long)(k0 + bk) * NC + (nb + bn);
      union { uint4 q; u16t u[8]; } r0, r1;
      r0.q = *(const uint4*)g0;
      r1.q = *(const uint4*)(g0 + NC);
      #pragma unroll
      for (int i = 0; i < 8; ++i) {
        u32t pack = (u32t)r0.u[i] | ((u32t)r1.u[i] << 16);
        *(u32t*)&sB[bn + i][bk] = pack;
      }
    }
    if (kk + 1 < ksteps) {
      __builtin_prefetch(A  + (long long)(mb + ar) * K       + (k0 + 32 + ah), 0, 1);
      __builtin_prefetch(Bp + (long long)(k0 + 32 + bk) * NC + (nb + bn),      0, 1);
    }
    __syncthreads();

    FragBF aF[4], bF[2];
    #pragma unroll
    for (int ms = 0; ms < 4; ++ms) {
      const u16t* p = &sA[wm * 64 + ms * 16 + l16][koff];
      aF[ms].q[0] = *(const uint4*)(p);
      aF[ms].q[1] = *(const uint4*)(p + 16);
    }
    #pragma unroll
    for (int ns = 0; ns < 2; ++ns) {
      const u16t* p = &sB[wn * 32 + ns * 16 + l16][koff];
      bF[ns].q[0] = *(const uint4*)(p);
      bF[ns].q[1] = *(const uint4*)(p + 16);
    }
    #pragma unroll
    for (int ms = 0; ms < 4; ++ms)
      #pragma unroll
      for (int ns = 0; ns < 2; ++ns)
        acc[ms][ns] = __builtin_amdgcn_wmma_f32_16x16x32_bf16(
            false, aF[ms].bv, false, bF[ns].bv, (short)0, acc[ms][ns], false, false);
    __syncthreads();
  }

  const u16t* Ep = E ? (E + (long long)blockIdx.z * cStride) : nullptr;
  #pragma unroll
  for (int ms = 0; ms < 4; ++ms)
    #pragma unroll
    for (int ns = 0; ns < 2; ++ns)
      #pragma unroll
      for (int el = 0; el < 8; ++el) {
        int row = mb + wm * 64 + ms * 16 + el + half * 8;
        int col = nb + wn * 32 + ns * 16 + l16;
        float v = alpha * acc[ms][ns][el];
        if (Ep) v += ebeta * bf2f(Ep[(long long)row * NC + col]);
        if (addIdent && row == col) v -= 1.0f;
        Cp[(long long)row * NC + col] = f2bf(v);
      }
}

// ---------------------------------------------------------------------------
// Channel mix: ACC[j][o] (+)= sum_c V[j][c] * W[o][c]
// V: 196608 x 64 bf16 (rows contiguous). W: 64x64 f32. ACC: 196608 x 64 f32.
// WG tile 256x64 (8 waves: 4 along M x 2 along N), K=64 in two 32-chunks.
// ---------------------------------------------------------------------------
__global__ __launch_bounds__(256) void mix_wmma(
    const u16t* __restrict__ V, const float* __restrict__ W,
    float* __restrict__ ACC, int first)
{
  const int tid  = threadIdx.x;
  const int lane = tid & 31;
  const int wv   = tid >> 5;
  const int wm   = wv & 3;
  const int wn   = wv >> 2;
  const int half = lane >> 4;
  const int l16  = lane & 15;
  const int koff = half * 8;
  const long long rbase = (long long)blockIdx.x * 256 + wm * 64;

  v8f acc[4][2] = {};

  #pragma unroll
  for (int cs = 0; cs < 2; ++cs) {
    const int cb = cs * 32;
    FragBF aF[4], bF[2];
    #pragma unroll
    for (int ms = 0; ms < 4; ++ms) {
      const long long row = rbase + ms * 16 + l16;
      const u16t* p = V + row * 64 + cb + koff;
      aF[ms].q[0] = *(const uint4*)(p);
      aF[ms].q[1] = *(const uint4*)(p + 16);
    }
    #pragma unroll
    for (int ns = 0; ns < 2; ++ns) {
      const int o = wn * 32 + ns * 16 + l16;
      const float* wp = W + o * 64 + cb + koff;
      #pragma unroll
      for (int i = 0; i < 8; ++i) {
        bF[ns].u[i]     = f2bf(wp[i]);
        bF[ns].u[8 + i] = f2bf(wp[16 + i]);
      }
    }
    #pragma unroll
    for (int ms = 0; ms < 4; ++ms)
      #pragma unroll
      for (int ns = 0; ns < 2; ++ns)
        acc[ms][ns] = __builtin_amdgcn_wmma_f32_16x16x32_bf16(
            false, aF[ms].bv, false, bF[ns].bv, (short)0, acc[ms][ns], false, false);
  }

  #pragma unroll
  for (int ms = 0; ms < 4; ++ms)
    #pragma unroll
    for (int ns = 0; ns < 2; ++ns)
      #pragma unroll
      for (int el = 0; el < 8; ++el) {
        long long row = rbase + ms * 16 + el + half * 8;
        int col = wn * 32 + ns * 16 + l16;
        long long idx = row * 64 + col;
        float v = acc[ms][ns][el];
        if (first) ACC[idx] = v;
        else       ACC[idx] += v;
      }
}

// ---------------------------------------------------------------------------
// Small preamble kernels (negligible cost, plain f32 VALU)
// ---------------------------------------------------------------------------
__global__ __launch_bounds__(256) void prep_small(
    const float* __restrict__ W1, const float* __restrict__ W2,
    const float* __restrict__ emb, const float* __restrict__ mlpw,
    float* __restrict__ e1n, float* __restrict__ e2n, float* __restrict__ Weff)
{
  __shared__ float s1[40][20], s2[40][20];
  const int tid = threadIdx.x;
  if (tid < 20) {                        // column softmax of W1
    float mx = -1e30f;
    for (int i = 0; i < 40; ++i) mx = fmaxf(mx, W1[i * 20 + tid]);
    float sm = 0.f;
    for (int i = 0; i < 40; ++i) sm += expf(W1[i * 20 + tid] - mx);
    for (int i = 0; i < 40; ++i) s1[i][tid] = expf(W1[i * 20 + tid] - mx) / sm;
  } else if (tid >= 32 && tid < 52) {    // column softmax of W2
    const int j = tid - 32;
    float mx = -1e30f;
    for (int i = 0; i < 40; ++i) mx = fmaxf(mx, W2[i * 20 + j]);
    float sm = 0.f;
    for (int i = 0; i < 40; ++i) sm += expf(W2[i * 20 + j] - mx);
    for (int i = 0; i < 40; ++i) s2[i][j] = expf(W2[i * 20 + j] - mx) / sm;
  }
  __syncthreads();
  for (int v = tid; v < 512; v += 256) {  // e1 = emb@s1, e2 = emb@s2, row-normalize
    float a1[20], a2[20];
    for (int j = 0; j < 20; ++j) { a1[j] = 0.f; a2[j] = 0.f; }
    for (int i = 0; i < 40; ++i) {
      const float e = emb[v * 40 + i];
      for (int j = 0; j < 20; ++j) { a1[j] += e * s1[i][j]; a2[j] += e * s2[i][j]; }
    }
    float n1 = 0.f, n2 = 0.f;
    for (int j = 0; j < 20; ++j) { n1 += a1[j] * a1[j]; n2 += a2[j] * a2[j]; }
    n1 = sqrtf(n1) + 1e-8f; n2 = sqrtf(n2) + 1e-8f;
    for (int j = 0; j < 20; ++j) {
      e1n[v * 20 + j] = a1[j] / n1;
      e2n[v * 20 + j] = a2[j] / n2;
    }
  }
  // Effective weight blocks: block0 = mlp blocks {0..3} summed (identity path),
  // blocks 1..9 = mlp blocks 4..12 (Ls,T2,T3 x powers 1..3)
  for (int idx = tid; idx < 10 * 4096; idx += 256) {
    const int k = idx >> 12, rem = idx & 4095, o = rem >> 6, c = rem & 63;
    float w;
    if (k == 0)
      w = mlpw[o * 832 + c] + mlpw[o * 832 + 64 + c] +
          mlpw[o * 832 + 128 + c] + mlpw[o * 832 + 192 + c];
    else
      w = mlpw[o * 832 + (3 + k) * 64 + c];
    Weff[idx] = w;
  }
}

__global__ __launch_bounds__(256) void rowsum_d(
    const float* __restrict__ adj, const float* __restrict__ e1n,
    const float* __restrict__ e2n, float* __restrict__ dvec)
{
  __shared__ float se1[20];
  __shared__ float red[256];
  const int v = blockIdx.x, tid = threadIdx.x;
  if (tid < 20) se1[tid] = e1n[v * 20 + tid];
  __syncthreads();
  float p = 0.f;
  for (int w = tid; w < 512; w += 256) {
    const float a = adj[v * 512 + w];
    if (a > 0.f) {
      float dot = 0.f;
      for (int j = 0; j < 20; ++j) dot += se1[j] * e2n[w * 20 + j];
      p += dot + a;
    } else p += 9e-15f;
  }
  red[tid] = p; __syncthreads();
  for (int s = 128; s > 0; s >>= 1) {
    if (tid < s) red[tid] += red[tid + s];
    __syncthreads();
  }
  if (tid == 0) dvec[v] = 1.0f / sqrtf(red[0]);
}

// S1[w][v] = Ls[v][w] = -(d[v] * A[v][w] * d[w])   (stored transposed, bf16)
__global__ __launch_bounds__(256) void build_s1(
    const float* __restrict__ adj, const float* __restrict__ e1n,
    const float* __restrict__ e2n, const float* __restrict__ dvec,
    u16t* __restrict__ S1)
{
  __shared__ float se2[20];
  const int w = blockIdx.x, tid = threadIdx.x;
  if (tid < 20) se2[tid] = e2n[w * 20 + tid];
  __syncthreads();
  const float dw = dvec[w];
  for (int v = tid; v < 512; v += 256) {
    const float a = adj[v * 512 + w];
    float Av;
    if (a > 0.f) {
      float dot = 0.f;
      for (int j = 0; j < 20; ++j) dot += se2[j] * e1n[v * 20 + j];
      Av = dot + a;
    } else Av = 9e-15f;
    S1[w * 512 + v] = f2bf(-(dvec[v] * Av * dw));
  }
}

__global__ void convert_to_bf16(const float* __restrict__ x, u16t* __restrict__ y,
                                long long n)
{
  long long i = (long long)blockIdx.x * 256 + threadIdx.x;
  const long long stride = (long long)gridDim.x * 256;
  for (; i < n; i += stride) y[i] = f2bf(x[i]);
}

__global__ void final_out(const float* __restrict__ ACC, const float* __restrict__ x,
                          const float* __restrict__ bias, float* __restrict__ out,
                          long long n)
{
  long long i = (long long)blockIdx.x * 256 + threadIdx.x;
  const long long stride = (long long)gridDim.x * 256;
  for (; i < n; i += stride)
    out[i] = fmaxf(ACC[i] + x[i] + bias[i & 63], 0.f);
}

// ---------------------------------------------------------------------------
extern "C" void kernel_launch(void* const* d_in, const int* in_sizes, int n_in,
                              void* d_out, int out_size, void* d_ws, size_t ws_size,
                              hipStream_t stream)
{
  (void)in_sizes; (void)n_in; (void)out_size; (void)ws_size;
  const float* x    = (const float*)d_in[0];
  const float* adj  = (const float*)d_in[1];
  const float* emb  = (const float*)d_in[2];
  const float* W1   = (const float*)d_in[3];
  const float* W2   = (const float*)d_in[4];
  const float* mlpw = (const float*)d_in[5];
  const float* mlpb = (const float*)d_in[6];
  float* out = (float*)d_out;

  char* ws = (char*)d_ws;
  size_t off = 0;
  auto take = [&](size_t bytes) -> char* {
    char* p = ws + off;
    off += (bytes + 255) & ~(size_t)255;
    return p;
  };
  const long long XE = 32LL * 512 * 768;          // bf16 elements per full tensor
  u16t* Xbf = (u16t*)take(XE * 2);
  u16t* Ua  = (u16t*)take(XE * 2);
  u16t* Ub  = (u16t*)take(XE * 2);
  u16t* S[3];
  S[0] = (u16t*)take(512 * 512 * 2);
  S[1] = (u16t*)take(512 * 512 * 2);
  S[2] = (u16t*)take(512 * 512 * 2);
  float* e1n  = (float*)take(512 * 20 * 4);
  float* e2n  = (float*)take(512 * 20 * 4);
  float* dv   = (float*)take(512 * 4);
  float* Weff = (float*)take(10 * 64 * 64 * 4);
  float* ACC  = (float*)take(32LL * 512 * 12 * 64 * 4);

  const long long NTOT = 32LL * 512 * 12 * 64;

  convert_to_bf16<<<2048, 256, 0, stream>>>(x, Xbf, NTOT);
  prep_small<<<1, 256, 0, stream>>>(W1, W2, emb, mlpw, e1n, e2n, Weff);
  rowsum_d<<<512, 256, 0, stream>>>(adj, e1n, e2n, dv);
  build_s1<<<512, 256, 0, stream>>>(adj, e1n, e2n, dv, S[0]);

  // S2 = 2*S1@S1 - I          (= T2^T, since S1 = Ls^T)
  gemm_bf16_wmma<<<dim3(4, 4, 1), 256, 0, stream>>>(
      S[0], S[0], S[1], 512, 512, 0, 0, 2.0f, nullptr, 0.f, 1);
  // S3 = 2*S1@S2 - S1         (= T3^T; S1 and S2 commute)
  gemm_bf16_wmma<<<dim3(4, 4, 1), 256, 0, stream>>>(
      S[0], S[1], S[2], 512, 512, 0, 0, 2.0f, S[0], -1.0f, 0);

  // identity path: ACC = Weff0 applied to x   (first=1 initializes ACC)
  mix_wmma<<<768, 256, 0, stream>>>(Xbf, Weff, ACC, 1);

  const long long BS = 512LL * 768;
  for (int s = 0; s < 3; ++s) {
    const u16t* Vsrc = Xbf;
    for (int p = 1; p <= 3; ++p) {
      u16t* Udst = (p & 1) ? Ua : Ub;
      gemm_bf16_wmma<<<dim3(6, 4, 32), 256, 0, stream>>>(
          S[s], Vsrc, Udst, 512, 768, BS, BS, 1.0f, nullptr, 0.f, 0);
      mix_wmma<<<768, 256, 0, stream>>>(
          Udst, Weff + (size_t)(1 + s * 3 + (p - 1)) * 4096, ACC, 0);
      Vsrc = Udst;
    }
  }

  final_out<<<2048, 256, 0, stream>>>(ACC, x, mlpb, out, NTOT);
}
